// PEN_69329362092470
// MI455X (gfx1250) — matrix operand order repaired and checked by reading, not compiled, not run on hardware
//
#include <hip/hip_runtime.h>

typedef __attribute__((ext_vector_type(16))) _Float16 v16h;
typedef __attribute__((ext_vector_type(2)))  _Float16 v2h;
typedef __attribute__((ext_vector_type(8)))  float    v8f;

static __device__ __forceinline__ float sigm(float x){ return 1.0f/(1.0f + __expf(-x)); }

// ---- WMMA fragment gather helpers (contiguous k = 16*(lane/16) + j, same order for A and B) ----
static __device__ __forceinline__ v16h loadAfrag(const _Float16* base, int lane, int rs){
  v16h a;
  int m    = lane & 15;
  int khi  = (lane < 16) ? 0  : 16;
  int khi2 = (lane < 16) ? 16 : 24;
#pragma unroll
  for (int j = 0; j < 16; ++j){
    int vg = j >> 1, pos = j & 1;
    int k = khi + j; (void)vg; (void)pos; (void)khi2;
    a[j] = base[m*rs + k];
  }
  return a;
}
static __device__ __forceinline__ v16h loadBfrag(const _Float16* base, int lane, int rs){
  v16h b;
  int n  = lane & 15;
  int kb = (lane < 16) ? 0 : 16;
#pragma unroll
  for (int j = 0; j < 16; ++j) b[j] = base[n*rs + kb + j];
  return b;
}
static __device__ __forceinline__ v8f wmma16(v16h a, v16h b, v8f c){
  return __builtin_amdgcn_wmma_f32_16x16x32_f16(false, a, false, b, (short)0, c, false, false);
}
static __device__ __forceinline__ void pack2(_Float16* dst, float v0, float v1){
  v2h p; p.x = (_Float16)v0; p.y = (_Float16)v1;
  *(v2h*)dst = p;   // 4B-aligned by construction (even column, row stride mult of 4B)
}

// ---- gfx1250 async global->LDS copy (16B per lane) ----
static __device__ __forceinline__ void async_ld16(unsigned lds_off, const void* gptr){
  asm volatile("global_load_async_to_lds_b128 %0, %1, off"
               :: "v"(lds_off), "v"((unsigned long long)(size_t)gptr)
               : "memory");
}
static __device__ __forceinline__ void async_wait0(){
  asm volatile("s_wait_asynccnt 0x0" ::: "memory");
}

// ============================================================================
// Gather-GEMM (im2col conv as GEMM): C[M,N] = A[M,K] * B
//   B[k,n] = X[b*bstr + ci*cstr + (t+kk)*tstr], k=(ci,kk), n=(b,t)
// Branch-free staging: clamped unconditional loads + select, paired b32 LDS stores.
// ============================================================================
__global__ void conv_gemm(const float* __restrict__ A, const float* __restrict__ X,
                          int bstr, int cstr, int tstr, int ks, int Tout,
                          float* __restrict__ C, int M, int N, int K)
{
  __shared__ _Float16 As[64][34];
  __shared__ _Float16 Bs[64][34];
  const int tid = threadIdx.x, lane = tid & 31, w = tid >> 5;
  const int blockM = blockIdx.y * 64, blockN = blockIdx.x * 64;
  const int mi  = w & 3;
  const int nj0 = (w >> 2) * 2;
  v8f acc0 = {}, acc1 = {};
  const int ksteps = (K + 31) >> 5;
  for (int kt = 0; kt < ksteps; ++kt){
    if (kt + 1 < ksteps)
      __builtin_prefetch(&A[(size_t)(blockM + (tid >> 5)) * K + (kt+1)*32], 0, 1);
    // stage A: 1024 half-pairs
#pragma unroll
    for (int p = 0; p < 4; ++p){
      int idx = tid + p*256;
      int m = idx >> 4, kp = (idx & 15) << 1;
      int gmr = blockM + m;
      int gm  = (gmr < M) ? gmr : (M - 1);
      int gk0 = kt*32 + kp, gk1 = gk0 + 1;
      float v0 = A[(size_t)gm*K + ((gk0 < K) ? gk0 : (K-1))];
      float v1 = A[(size_t)gm*K + ((gk1 < K) ? gk1 : (K-1))];
      if (gk0 >= K || gmr >= M) v0 = 0.f;
      if (gk1 >= K || gmr >= M) v1 = 0.f;
      pack2(&As[m][kp], v0, v1);
    }
    // stage B (im2col gather)
#pragma unroll
    for (int p = 0; p < 4; ++p){
      int idx = tid + p*256;
      int n = idx >> 4, kp = (idx & 15) << 1;
      int gn = blockN + n;                 // N is multiple of 64 for all launches
      int b  = gn / Tout, t = gn - b*Tout;
      int gk0 = kt*32 + kp, gk1 = gk0 + 1;
      int k0 = (gk0 < K) ? gk0 : (K-1);
      int k1 = (gk1 < K) ? gk1 : (K-1);
      int ci0 = k0 / ks, kk0 = k0 - ci0*ks;
      int ci1 = k1 / ks, kk1 = k1 - ci1*ks;
      float v0 = X[(size_t)b*bstr + (size_t)ci0*cstr + (size_t)(t+kk0)*tstr];
      float v1 = X[(size_t)b*bstr + (size_t)ci1*cstr + (size_t)(t+kk1)*tstr];
      if (gk0 >= K) v0 = 0.f;
      if (gk1 >= K) v1 = 0.f;
      pack2(&Bs[n][kp], v0, v1);
    }
    __syncthreads();
    v16h af = loadAfrag(&As[mi*16][0], lane, 34);
    v16h b0 = loadBfrag(&Bs[nj0*16][0], lane, 34);
    v16h b1 = loadBfrag(&Bs[(nj0+1)*16][0], lane, 34);
    acc0 = wmma16(af, b0, acc0);
    acc1 = wmma16(af, b1, acc1);
    __syncthreads();
  }
  const int mhi = (lane < 16) ? 0 : 8;
#pragma unroll
  for (int r = 0; r < 8; ++r){
    int m  = blockM + mi*16 + r + mhi;
    int n0 = blockN + nj0*16 + (lane & 15);
    if (m < M){
      C[(size_t)m*N + n0]      = acc0[r];
      C[(size_t)m*N + n0 + 16] = acc1[r];
    }
  }
}

// ============================================================================
// BatchNorm (training-mode batch stats) + LeakyReLU, feature map stored [C][Nn]
// ============================================================================
__global__ void bn_stats(const float* __restrict__ d, int Nn,
                         float* __restrict__ meanO, float* __restrict__ istdO)
{
  __shared__ float s1[256], s2[256];
  const int c = blockIdx.x, tid = threadIdx.x;
  const float* p = d + (size_t)c*Nn;
  float a = 0.f, b = 0.f;
  for (int i = tid; i < Nn; i += 256){ float v = p[i]; a += v; b += v*v; }
  s1[tid] = a; s2[tid] = b; __syncthreads();
  for (int o = 128; o > 0; o >>= 1){
    if (tid < o){ s1[tid] += s1[tid+o]; s2[tid] += s2[tid+o]; }
    __syncthreads();
  }
  if (tid == 0){
    float m = s1[0] / (float)Nn;
    float var = s2[0] / (float)Nn - m*m;
    meanO[c] = m; istdO[c] = rsqrtf(var + 1e-5f);
  }
}

__global__ void bn_lrelu(float* __restrict__ d, int Nn,
                         const float* __restrict__ mean, const float* __restrict__ istd,
                         const float* __restrict__ g, const float* __restrict__ bta)
{
  const int c = blockIdx.y;
  const int i = blockIdx.x*256 + threadIdx.x;
  if (i < Nn){
    size_t idx = (size_t)c*Nn + i;
    float v = g[c]*(d[idx] - mean[c])*istd[c] + bta[c];
    d[idx] = (v >= 0.f) ? v : 0.01f*v;
  }
}

// ============================================================================
// Weight-stationary WMMA LSTM scan. One block = 32 batches, full T loop.
// Emits hidden sequence + finals in f16 (consumed by async-staged attention).
// grid = (B/32, 3): blockIdx.y selects which of the q/k/v LSTMs.
// ============================================================================
template<int KIN, int KPAD>
__global__ void lstm_cells(const float* __restrict__ xin, int xbs, int xcs, int xts,
                           const float* Wih0, const float* Whh0, const float* bi0, const float* bh0,
                           const float* Wih1, const float* Whh1, const float* bi1, const float* bh1,
                           const float* Wih2, const float* Whh2, const float* bi2, const float* bh2,
                           const float* __restrict__ h0, const float* __restrict__ c0,
                           _Float16* out0, _Float16* out1, _Float16* out2,
                           _Float16* hfin, _Float16* cfin, int T)
{
  constexpr int KTOT = 64 + KPAD;
  constexpr int KST  = KTOT / 32;
  constexpr int BB = 32, H = 64, NG = 256;
  constexpr int XIT = (BB*KIN + 255) / 256;

  __shared__ _Float16 hA[BB][KTOT + 2];   // [h(64) | x(KIN) | zero-pad]
  __shared__ float    gbuf[BB][NG];
  __shared__ float    cbuf[BB][H];
  __shared__ float    bias[NG];

  const int tid = threadIdx.x, lane = tid & 31, w = tid >> 5;
  const int y = blockIdx.y;
  const float* Wih = (y==0) ? Wih0 : ((y==1) ? Wih1 : Wih2);
  const float* Whh = (y==0) ? Whh0 : ((y==1) ? Whh1 : Whh2);
  const float* bi  = (y==0) ? bi0  : ((y==1) ? bi1  : bi2);
  const float* bh  = (y==0) ? bh0  : ((y==1) ? bh1  : bh2);
  _Float16* outSeq = (y==0) ? out0 : ((y==1) ? out1 : out2);
  _Float16* hf = hfin + (size_t)y * 1024 * H;
  _Float16* cf = cfin + (size_t)y * 1024 * H;
  const int b0 = blockIdx.x * BB;

  bias[tid] = bi[tid] + bh[tid];
  for (int idx = tid; idx < BB*H; idx += 256){
    int bb = idx >> 6, j = idx & 63;
    hA[bb][j]   = (_Float16)h0[(size_t)(b0+bb)*H + j];
    cbuf[bb][j] = c0[(size_t)(b0+bb)*H + j];
  }
  if (KPAD > KIN){
    constexpr int PADN = (KPAD > KIN) ? (KPAD - KIN) : 1;
    for (int idx = tid; idx < BB*PADN; idx += 256){
      int bb = idx / PADN, r = idx - bb*PADN;
      hA[bb][64 + KIN + r] = (_Float16)0.f;
    }
  }

  // Stationary weight fragments in registers (branch-free clamped gather):
  // B[k,n] = (k<64) ? Whh[n][k] : Wih[n][k-64]  (zero pad beyond)
  const int njBase = (w >> 1) * 4;
  const int miT    = (w & 1);
  v16h bf[4][KST];
#pragma unroll
  for (int nt = 0; nt < 4; ++nt){
    int nb = (njBase + nt)*16 + (lane & 15);
#pragma unroll
    for (int kt = 0; kt < KST; ++kt){
      int kb = ((lane < 16) ? 0 : 16) + kt*32;
#pragma unroll
      for (int j = 0; j < 16; ++j){
        int k  = kb + j;
        int kH = (k < 64) ? k : 63;
        int kI = k - 64; kI = (kI < 0) ? 0 : ((kI >= KIN) ? (KIN-1) : kI);
        float vh = Whh[(size_t)nb*64  + kH];
        float vi = Wih[(size_t)nb*KIN + kI];
        float v  = (k < 64) ? vh : ((k < 64 + KIN) ? vi : 0.f);
        bf[nt][kt][j] = (_Float16)v;
      }
    }
  }
  __syncthreads();

  for (int t = 0; t < T; ++t){
    // stage x_t (unconditional clamped loads; only the store is guarded)
#pragma unroll
    for (int s = 0; s < XIT; ++s){
      int idx = tid + s*256;
      int idc = (idx < BB*KIN) ? idx : (BB*KIN - 1);
      int bb = idc / KIN, ci = idc - bb*KIN;
      float v = xin[(size_t)(b0+bb)*xbs + (size_t)ci*xcs + (size_t)t*xts];
      if (idx < BB*KIN) hA[bb][64 + ci] = (_Float16)v;
    }
    __syncthreads();

    v8f acc[4];
#pragma unroll
    for (int nt = 0; nt < 4; ++nt){
#pragma unroll
      for (int r = 0; r < 8; ++r) acc[nt][r] = 0.0f;
    }
    v16h af[KST];
#pragma unroll
    for (int kt = 0; kt < KST; ++kt)
      af[kt] = loadAfrag(&hA[miT*16][kt*32], lane, KTOT + 2);
#pragma unroll
    for (int nt = 0; nt < 4; ++nt){
#pragma unroll
      for (int kt = 0; kt < KST; ++kt)
        acc[nt] = wmma16(af[kt], bf[nt][kt], acc[nt]);
    }
    const int mhi = (lane < 16) ? 0 : 8;
#pragma unroll
    for (int nt = 0; nt < 4; ++nt){
      int n = (njBase + nt)*16 + (lane & 15);
#pragma unroll
      for (int r = 0; r < 8; ++r)
        gbuf[miT*16 + r + mhi][n] = acc[nt][r];
    }
    __syncthreads();

    // cell update (torch gate order i,f,g,o)
    for (int idx = tid; idx < BB*H; idx += 256){
      int bb = idx >> 6, j = idx & 63;
      float gi = gbuf[bb][j]       + bias[j];
      float gf = gbuf[bb][64 + j]  + bias[64 + j];
      float gg = gbuf[bb][128 + j] + bias[128 + j];
      float go = gbuf[bb][192 + j] + bias[192 + j];
      float cn = sigm(gf)*cbuf[bb][j] + sigm(gi)*tanhf(gg);
      float hn = sigm(go)*tanhf(cn);
      cbuf[bb][j] = cn;
      hA[bb][j]   = (_Float16)hn;
      outSeq[((size_t)(b0+bb)*T + t)*H + j] = (_Float16)hn;
      if (t == T-1){
        hf[(size_t)(b0+bb)*H + j] = (_Float16)hn;
        cf[(size_t)(b0+bb)*H + j] = (_Float16)cn;
      }
    }
    __syncthreads();
  }
}

// ============================================================================
// Batched scores: S[b] = Q[b] @ K[b]^T.  Q,K are f16 rows of 64, contiguous.
// Tiles staged via gfx1250 GLOBAL_LOAD_ASYNC_TO_LDS_B128 (no conversion needed).
// ============================================================================
__global__ void attn_scores(const _Float16* __restrict__ Q, const _Float16* __restrict__ Kk,
                            float* __restrict__ S, int N,
                            long long bsA, long long bsB, long long bsC)
{
  __shared__ alignas(16) _Float16 As[64][72];   // 144B row stride: 16B-aligned chunks
  __shared__ alignas(16) _Float16 Bs[64][72];
  const int tid = threadIdx.x, lane = tid & 31, w = tid >> 5;
  const int b = blockIdx.z;
  const _Float16* Qb = Q + (size_t)b*bsA;
  const _Float16* Kb = Kk + (size_t)b*bsB;
  float* Sb = S + (size_t)b*bsC;
  const int blockM = blockIdx.y*64, blockN = blockIdx.x*64;

  const unsigned asBase = (unsigned)(size_t)&As[0][0];
  const unsigned bsBase = (unsigned)(size_t)&Bs[0][0];
#pragma unroll
  for (int r = 0; r < 2; ++r){
    int idx = tid + r*256;            // 512 x 16B chunks per tile
    int row = idx >> 3, c = idx & 7;
    async_ld16(asBase + row*144 + c*16, Qb + (size_t)(blockM+row)*64 + c*8);
    async_ld16(bsBase + row*144 + c*16, Kb + (size_t)(blockN+row)*64 + c*8);
  }
  async_wait0();
  __syncthreads();

  const int mi = w & 3, nj0 = (w >> 2) * 2;
  v8f acc0 = {}, acc1 = {};
#pragma unroll
  for (int kt = 0; kt < 2; ++kt){
    v16h a  = loadAfrag(&As[mi*16][kt*32], lane, 72);
    v16h b0 = loadBfrag(&Bs[nj0*16][kt*32], lane, 72);
    v16h b1 = loadBfrag(&Bs[(nj0+1)*16][kt*32], lane, 72);
    acc0 = wmma16(a, b0, acc0);
    acc1 = wmma16(a, b1, acc1);
  }
  const int mhi = (lane < 16) ? 0 : 8;
#pragma unroll
  for (int r = 0; r < 8; ++r){
    int m  = blockM + mi*16 + r + mhi;
    int n0 = blockN + nj0*16 + (lane & 15);
    Sb[(size_t)m*N + n0]      = acc0[r];
    Sb[(size_t)m*N + n0 + 16] = acc1[r];
  }
}

// ---- global softmax reductions (axis=None over all elements) ----
__global__ void red_part_max(const float* __restrict__ d, int n, float* __restrict__ part){
  __shared__ float s[256];
  const int tid = threadIdx.x;
  float m = -3.4e38f;
  for (int i = blockIdx.x*256 + tid; i < n; i += 256*256) m = fmaxf(m, d[i]);
  s[tid] = m; __syncthreads();
  for (int o = 128; o > 0; o >>= 1){ if (tid < o) s[tid] = fmaxf(s[tid], s[tid+o]); __syncthreads(); }
  if (tid == 0) part[blockIdx.x] = s[0];
}
__global__ void red_fin_max(const float* __restrict__ part, float* __restrict__ out){
  __shared__ float s[256];
  const int tid = threadIdx.x;
  s[tid] = part[tid]; __syncthreads();
  for (int o = 128; o > 0; o >>= 1){ if (tid < o) s[tid] = fmaxf(s[tid], s[tid+o]); __syncthreads(); }
  if (tid == 0) *out = s[0];
}
__global__ void red_part_expsum(const float* __restrict__ d, int n,
                                const float* __restrict__ gmax, float* __restrict__ part){
  __shared__ float s[256];
  const float gm = *gmax;
  const int tid = threadIdx.x;
  float a = 0.f;
  for (int i = blockIdx.x*256 + tid; i < n; i += 256*256) a += __expf(d[i] - gm);
  s[tid] = a; __syncthreads();
  for (int o = 128; o > 0; o >>= 1){ if (tid < o) s[tid] += s[tid+o]; __syncthreads(); }
  if (tid == 0) part[blockIdx.x] = s[0];
}
__global__ void red_fin_sum(const float* __restrict__ part, float* __restrict__ out){
  __shared__ float s[256];
  const int tid = threadIdx.x;
  s[tid] = part[tid]; __syncthreads();
  for (int o = 128; o > 0; o >>= 1){ if (tid < o) s[tid] += s[tid+o]; __syncthreads(); }
  if (tid == 0) *out = s[0];
}

// ============================================================================
// Apply: O[b] = softmax_global(S)[b] @ V[b].  p staged as f16*2^14; V is f16.
// ============================================================================
__global__ void attn_apply(const float* __restrict__ S, const _Float16* __restrict__ V,
                           float* __restrict__ O, int Kd,
                           long long bsS, long long bsV, long long bsO,
                           const float* __restrict__ gmaxp, const float* __restrict__ gsump)
{
  __shared__ _Float16 As[64][34];
  __shared__ _Float16 Bs[64][34];
  const int tid = threadIdx.x, lane = tid & 31, w = tid >> 5;
  const int b = blockIdx.z;
  const float* Sb = S + (size_t)b*bsS;
  const _Float16* Vb = V + (size_t)b*bsV;
  float* Ob = O + (size_t)b*bsO;
  const int blockM = blockIdx.y*64;       // N-block = 0 (N==64)
  const float gm = *gmaxp;
  const float inv = 1.0f / (*gsump);
  const float SCALE = 16384.0f;
  const int mi = w & 3, nj0 = (w >> 2) * 2;
  v8f acc0 = {}, acc1 = {};
  const int ksteps = Kd >> 5;
  for (int kt = 0; kt < ksteps; ++kt){
#pragma unroll
    for (int p = 0; p < 4; ++p){
      int idx = tid + p*256;
      int m = idx >> 4, kp = (idx & 15) << 1;
      float s0 = Sb[(size_t)(blockM+m)*Kd + kt*32 + kp];
      float s1 = Sb[(size_t)(blockM+m)*Kd + kt*32 + kp + 1];
      pack2(&As[m][kp], __expf(s0 - gm)*SCALE, __expf(s1 - gm)*SCALE);
      // BT[n][k] = V[k][n]  (strided f16 gather)
      v2h pv;
      pv.x = Vb[(size_t)(kt*32 + kp    )*64 + m];
      pv.y = Vb[(size_t)(kt*32 + kp + 1)*64 + m];
      *(v2h*)&Bs[m][kp] = pv;
    }
    __syncthreads();
    v16h a  = loadAfrag(&As[mi*16][0], lane, 34);
    v16h b0 = loadBfrag(&Bs[nj0*16][0], lane, 34);
    v16h b1 = loadBfrag(&Bs[(nj0+1)*16][0], lane, 34);
    acc0 = wmma16(a, b0, acc0);
    acc1 = wmma16(a, b1, acc1);
    __syncthreads();
  }
  const float scl = inv / SCALE;
  const int mhi = (lane < 16) ? 0 : 8;
#pragma unroll
  for (int r = 0; r < 8; ++r){
    int m  = blockM + mi*16 + r + mhi;
    int n0 = nj0*16 + (lane & 15);
    Ob[(size_t)m*64 + n0]      = acc0[r]*scl;
    Ob[(size_t)m*64 + n0 + 16] = acc1[r]*scl;
  }
}

// ============================================================================
// FC (virtual concat of rn2=att2^T and avg-pooled conv3) + row softmax
// ============================================================================
__global__ void fc_softmax(const float* __restrict__ att2, const float* __restrict__ f3,
                           const float* __restrict__ fcw, const float* __restrict__ fcb,
                           float* __restrict__ out)
{
  __shared__ float red[5][256];
  const int b = blockIdx.x, tid = threadIdx.x;
  float s[5] = {0.f,0.f,0.f,0.f,0.f};
  const int QTOT = 64*236;
  const size_t N3 = (size_t)1024*108;
  for (int q = tid; q < QTOT; q += 256){
    int j = q / 236, i = q - j*236;
    float v;
    if (i < 128){
      v = att2[((size_t)b*128 + i)*64 + j];                 // rn2[b][j][i] = att2[b][i][j]
    } else {
      int l = i - 128;                                       // avgpool over channel pairs
      v = 0.5f*(f3[(size_t)(2*j)*N3   + (size_t)b*108 + l] +
                f3[(size_t)(2*j+1)*N3 + (size_t)b*108 + l]);
    }
#pragma unroll
    for (int c = 0; c < 5; ++c) s[c] += fcw[(size_t)c*QTOT + q] * v;
  }
#pragma unroll
  for (int c = 0; c < 5; ++c) red[c][tid] = s[c];
  __syncthreads();
  for (int o = 128; o > 0; o >>= 1){
    if (tid < o){
#pragma unroll
      for (int c = 0; c < 5; ++c) red[c][tid] += red[c][tid+o];
    }
    __syncthreads();
  }
  if (tid == 0){
    float lg[5], mx = -3.4e38f;
#pragma unroll
    for (int c = 0; c < 5; ++c){ lg[c] = red[c][0] + fcb[c]; mx = fmaxf(mx, lg[c]); }
    float sm = 0.f;
#pragma unroll
    for (int c = 0; c < 5; ++c){ lg[c] = __expf(lg[c] - mx); sm += lg[c]; }
#pragma unroll
    for (int c = 0; c < 5; ++c) out[(size_t)b*5 + c] = lg[c]/sm;
  }
}

// ============================================================================
extern "C" void kernel_launch(void* const* d_in, const int* in_sizes, int n_in,
                              void* d_out, int out_size, void* d_ws, size_t ws_size,
                              hipStream_t stream) {
  (void)in_sizes; (void)n_in; (void)out_size; (void)ws_size;
  const float* x    = (const float*)d_in[0];
  const float* h0   = (const float*)d_in[1];
  const float* c0   = (const float*)d_in[2];
  const float* c1w  = (const float*)d_in[3];
  const float* bn1g = (const float*)d_in[5];  const float* bn1b = (const float*)d_in[6];
  const float* c2w  = (const float*)d_in[7];
  const float* bn2g = (const float*)d_in[9];  const float* bn2b = (const float*)d_in[10];
  const float* c3w  = (const float*)d_in[11];
  const float* bn3g = (const float*)d_in[13]; const float* bn3b = (const float*)d_in[14];
  const float* fcw  = (const float*)d_in[39];
  const float* fcb  = (const float*)d_in[40];
  float* out = (float*)d_out;
  float* wsf = (float*)d_ws;

  const int B = 1024, T = 128, H = 64;
  const int N1 = 1024*118, N2 = 1024*112, N3 = 1024*108;

  size_t o = 0;
  float* convA = wsf + o; o += (size_t)128*N1;        // conv1 out, later conv3 out (f3)
  float* convB = wsf + o; o += (size_t)128*N2;        // conv2 out
  _Float16* Qh  = (_Float16*)(wsf + o); o += (size_t)B*T*H/2;   // f16 seqs
  _Float16* Kh  = (_Float16*)(wsf + o); o += (size_t)B*T*H/2;
  _Float16* Vh  = (_Float16*)(wsf + o); o += (size_t)B*T*H/2;
  float* R      = wsf + o; o += (size_t)B*T*H;        // rn1, later att2 (f32)
  float* S      = wsf + o; o += (size_t)B*T*T;        // scores (f32)
  _Float16* HFh = (_Float16*)(wsf + o); o += (size_t)3*B*H/2;   // set1 h finals (f16)
  _Float16* CFh = (_Float16*)(wsf + o); o += (size_t)3*B*H/2;
  float* HO     = wsf + o; o += (size_t)B*H;          // h_out (f32)
  float* CO     = wsf + o; o += (size_t)B*H;          // c_out (f32)
  _Float16* HF2 = (_Float16*)(wsf + o); o += (size_t)3*B*H/2;   // set2 finals (sink)
  _Float16* CF2 = (_Float16*)(wsf + o); o += (size_t)3*B*H/2;
  float* REDp   = wsf + o; o += 256;
  float* GMAX   = wsf + o; o += 1;
  float* GSUM   = wsf + o; o += 1;
  float* MEAN   = wsf + o; o += 128;
  float* ISTD   = wsf + o; o += 128;

  // ---- conv stack (bias dropped: cancelled by training-mode BN mean) ----
  conv_gemm<<<dim3(N1/64, 2), 256, 0, stream>>>(c1w, x,     12*128, 128, 1, 11, 118, convA, 128, N1, 12*11);
  bn_stats <<<128, 256, 0, stream>>>(convA, N1, MEAN, ISTD);
  bn_lrelu <<<dim3((N1+255)/256, 128), 256, 0, stream>>>(convA, N1, MEAN, ISTD, bn1g, bn1b);

  conv_gemm<<<dim3(N2/64, 2), 256, 0, stream>>>(c2w, convA, 118, N1, 1, 7, 112, convB, 128, N2, 128*7);
  bn_stats <<<128, 256, 0, stream>>>(convB, N2, MEAN, ISTD);
  bn_lrelu <<<dim3((N2+255)/256, 128), 256, 0, stream>>>(convB, N2, MEAN, ISTD, bn2g, bn2b);

  conv_gemm<<<dim3(N3/64, 2), 256, 0, stream>>>(c3w, convB, 112, N2, 1, 5, 108, convA, 128, N3, 128*5);
  bn_stats <<<128, 256, 0, stream>>>(convA, N3, MEAN, ISTD);
  bn_lrelu <<<dim3((N3+255)/256, 128), 256, 0, stream>>>(convA, N3, MEAN, ISTD, bn3g, bn3b);

  // ---- LSTM set 1 (q,k,v): input x [B,CIN,T] ----
  lstm_cells<12,32><<<dim3(B/32, 3), 256, 0, stream>>>(
      x, 12*128, 128, 1,
      (const float*)d_in[15], (const float*)d_in[16], (const float*)d_in[17], (const float*)d_in[18],
      (const float*)d_in[19], (const float*)d_in[20], (const float*)d_in[21], (const float*)d_in[22],
      (const float*)d_in[23], (const float*)d_in[24], (const float*)d_in[25], (const float*)d_in[26],
      h0, c0, Qh, Kh, Vh, HFh, CFh, T);

  // ---- attention 1 (global softmax over all B*T*T scores) ----
  attn_scores<<<dim3(2, 2, B), 256, 0, stream>>>(Qh, Kh, S, T,
      (long long)T*H, (long long)T*H, (long long)T*T);
  {
    int n = B*T*T;
    red_part_max   <<<256, 256, 0, stream>>>(S, n, REDp);
    red_fin_max    <<<1,   256, 0, stream>>>(REDp, GMAX);
    red_part_expsum<<<256, 256, 0, stream>>>(S, n, GMAX, REDp);
    red_fin_sum    <<<1,   256, 0, stream>>>(REDp, GSUM);
  }
  attn_apply<<<dim3(1, 2, B), 256, 0, stream>>>(S, Vh, R, T,
      (long long)T*T, (long long)T*H, (long long)T*H, GMAX, GSUM);

  // ---- batch-mixing attention on final h states ----
  attn_scores<<<dim3(16, 16, 1), 256, 0, stream>>>(HFh, HFh + (size_t)B*H, S, B, 0, 0, 0);
  {
    int n = B*B;
    red_part_max   <<<256, 256, 0, stream>>>(S, n, REDp);
    red_fin_max    <<<1,   256, 0, stream>>>(REDp, GMAX);
    red_part_expsum<<<256, 256, 0, stream>>>(S, n, GMAX, REDp);
    red_fin_sum    <<<1,   256, 0, stream>>>(REDp, GSUM);
  }
  attn_apply<<<dim3(1, 16, 1), 256, 0, stream>>>(S, HFh + (size_t)2*B*H, HO, B, 0, 0, 0, GMAX, GSUM);

  // ---- batch-mixing attention on final c states ----
  attn_scores<<<dim3(16, 16, 1), 256, 0, stream>>>(CFh, CFh + (size_t)B*H, S, B, 0, 0, 0);
  {
    int n = B*B;
    red_part_max   <<<256, 256, 0, stream>>>(S, n, REDp);
    red_fin_max    <<<1,   256, 0, stream>>>(REDp, GMAX);
    red_part_expsum<<<256, 256, 0, stream>>>(S, n, GMAX, REDp);
    red_fin_sum    <<<1,   256, 0, stream>>>(REDp, GSUM);
  }
  attn_apply<<<dim3(1, 16, 1), 256, 0, stream>>>(S, CFh + (size_t)2*B*H, CO, B, 0, 0, 0, GMAX, GSUM);

  // ---- LSTM set 2 (q1,k1,v1): input rn1 [B,T,H] (f32), init (h_out, c_out) ----
  lstm_cells<64,64><<<dim3(B/32, 3), 256, 0, stream>>>(
      R, T*H, 1, H,
      (const float*)d_in[27], (const float*)d_in[28], (const float*)d_in[29], (const float*)d_in[30],
      (const float*)d_in[31], (const float*)d_in[32], (const float*)d_in[33], (const float*)d_in[34],
      (const float*)d_in[35], (const float*)d_in[36], (const float*)d_in[37], (const float*)d_in[38],
      HO, CO, Qh, Kh, Vh, HF2, CF2, T);

  // ---- attention 2 ----
  attn_scores<<<dim3(2, 2, B), 256, 0, stream>>>(Qh, Kh, S, T,
      (long long)T*H, (long long)T*H, (long long)T*T);
  {
    int n = B*T*T;
    red_part_max   <<<256, 256, 0, stream>>>(S, n, REDp);
    red_fin_max    <<<1,   256, 0, stream>>>(REDp, GMAX);
    red_part_expsum<<<256, 256, 0, stream>>>(S, n, GMAX, REDp);
    red_fin_sum    <<<1,   256, 0, stream>>>(REDp, GSUM);
  }
  attn_apply<<<dim3(1, 2, B), 256, 0, stream>>>(S, Vh, R, T,
      (long long)T*T, (long long)T*H, (long long)T*H, GMAX, GSUM);

  // ---- FC (virtual concat) + softmax ----
  fc_softmax<<<B, 256, 0, stream>>>(R, convA, fcw, fcb, out);
}
